// LLMAttention_26499948216612
// MI455X (gfx1250) — matrix operand order repaired and checked by reading, not compiled, not run on hardware
//
#include <hip/hip_runtime.h>
#include <math.h>

// ---------- types ----------
typedef __bf16 bf16;
typedef float v8f  __attribute__((ext_vector_type(8)));
typedef bf16  v16bf __attribute__((ext_vector_type(16)));
typedef bf16  v8bf  __attribute__((ext_vector_type(8)));

union FragU { v16bf v; v8bf h[2]; };

static __device__ __forceinline__ v8f wmma_bf16(const FragU& a, const FragU& b, v8f c) {
    return __builtin_amdgcn_wmma_f32_16x16x32_bf16(false, a.v, false, b.v,
                                                   (short)0, c, false, false);
}

// Async global -> LDS copy (CDNA5, ASYNCcnt). LDS address = low 32 bits of the
// generic pointer (ISA: LDS aperture address truncates to addr[31:0]).
static __device__ __forceinline__ void async_b128(void* lds_dst, const void* gsrc) {
    unsigned lds = (unsigned)(size_t)lds_dst;
    asm volatile("global_load_async_to_lds_b128 %0, %1, off"
                 :: "v"(lds), "v"(gsrc) : "memory");
}
static __device__ __forceinline__ void wait_async0() {
    asm volatile("s_wait_asynccnt 0x0" ::: "memory");
}

// ---------- constants ----------
#define BB 2
#define TT 2048
#define DD 2048
#define HH 16
#define HD 128

// ======================================================================
// cast f32 -> bf16
// ======================================================================
__global__ __launch_bounds__(256)
void cast_f32_bf16(const float* __restrict__ in, bf16* __restrict__ out, int n) {
    int i = blockIdx.x * 256 + threadIdx.x;
    if (i < n) out[i] = (bf16)in[i];
}

// ======================================================================
// GEMM: C[M,N] = A[M,K] * B[N,K]^T  (A,B bf16 row-major, C f32)
// block tile 128x128, K-step 32, 8 waves (4x2), wave tile 32x64
// double-buffered LDS, tiles staged with global_load_async_to_lds_b128
// ======================================================================
__global__ __launch_bounds__(256)
void gemm_bf16_wmma(const bf16* __restrict__ A, const bf16* __restrict__ Bw,
                    float* __restrict__ C, int M, int N, int K) {
    __shared__ bf16 As[2][128][32];
    __shared__ bf16 Bs[2][128][32];

    const int tid   = threadIdx.x;
    const int lane  = tid & 31;
    const int wave  = tid >> 5;
    const int l15   = lane & 15;
    const int half  = lane >> 4;           // 0: lanes 0-15, 1: lanes 16-31
    const int mBlk  = blockIdx.y * 128;
    const int nBlk  = blockIdx.x * 128;
    const int wm    = (wave >> 1) * 32;    // 0,32,64,96
    const int wn    = (wave & 1) * 64;     // 0,64

    v8f acc[2][4];
    for (int i = 0; i < 2; ++i)
        for (int j = 0; j < 4; ++j) acc[i][j] = {};

    const int ldRow = tid >> 1;            // 0..127
    const int ldCol = (tid & 1) * 16;      // 0 or 16
    const int kbA   = half * 8;            // A-frag K base per ISA layout
    const int kbB   = half * 8;            // B-frag K base per ISA layout

    // prologue: stage tile 0 into buffer 0
    {
        const bf16* ag = A  + (size_t)(mBlk + ldRow) * K + ldCol;
        async_b128(&As[0][ldRow][ldCol],     ag);
        async_b128(&As[0][ldRow][ldCol + 8], ag + 8);
        const bf16* bg = Bw + (size_t)(nBlk + ldRow) * K + ldCol;
        async_b128(&Bs[0][ldRow][ldCol],     bg);
        async_b128(&Bs[0][ldRow][ldCol + 8], bg + 8);
    }

    int buf = 0;
    for (int k0 = 0; k0 < K; k0 += 32) {
        wait_async0();        // our async writes into buf are done
        __syncthreads();      // everyone's writes done, prior reads of buf^1 done

        if (k0 + 32 < K) {    // stage next tile into the other buffer (overlaps WMMA)
            const int nb = buf ^ 1;
            const bf16* ag = A  + (size_t)(mBlk + ldRow) * K + (k0 + 32) + ldCol;
            async_b128(&As[nb][ldRow][ldCol],     ag);
            async_b128(&As[nb][ldRow][ldCol + 8], ag + 8);
            const bf16* bg = Bw + (size_t)(nBlk + ldRow) * K + (k0 + 32) + ldCol;
            async_b128(&Bs[nb][ldRow][ldCol],     bg);
            async_b128(&Bs[nb][ldRow][ldCol + 8], bg + 8);
        }

        FragU af[2], bfr[4];
        for (int i = 0; i < 2; ++i) {
            int r = wm + i * 16 + l15;
            af[i].h[0] = *(const v8bf*)&As[buf][r][kbA];
            af[i].h[1] = *(const v8bf*)&As[buf][r][kbA + 16];
        }
        for (int j = 0; j < 4; ++j) {
            int c = wn + j * 16 + l15;
            bfr[j].h[0] = *(const v8bf*)&Bs[buf][c][kbB];
            bfr[j].h[1] = *(const v8bf*)&Bs[buf][c][kbB + 16];
        }
        for (int i = 0; i < 2; ++i)
            for (int j = 0; j < 4; ++j)
                acc[i][j] = wmma_bf16(af[i], bfr[j], acc[i][j]);

        buf ^= 1;
    }

    for (int i = 0; i < 2; ++i)
        for (int j = 0; j < 4; ++j) {
            int rowBase = mBlk + wm + i * 16 + half * 8;
            int col     = nBlk + wn + j * 16 + l15;
            for (int r = 0; r < 8; ++r)
                C[(size_t)(rowBase + r) * N + col] = acc[i][j][r];
        }
}

// ======================================================================
// RMSNorm (over HD=128) + interleaved RoPE; [b,t,h,hd] f32 -> [b,h,t,hd] bf16
// one 128-thread block per (b,t,h)
// ======================================================================
__global__ __launch_bounds__(128)
void rmsnorm_rope(const float* __restrict__ in, const float* __restrict__ w,
                  bf16* __restrict__ out) {
    __shared__ float red[4];
    const int idx = blockIdx.x;
    const int h = idx % HH;
    const int t = (idx / HH) % TT;
    const int b = idx / (HH * TT);
    const int tid = threadIdx.x;

    float v = in[((size_t)((size_t)b * TT + t) * HH + h) * HD + tid];
    float sq = v * v;
    for (int m = 16; m >= 1; m >>= 1) sq += __shfl_xor(sq, m, 32);
    if ((tid & 31) == 0) red[tid >> 5] = sq;
    __syncthreads();
    float tot = red[0] + red[1] + red[2] + red[3];
    float rn  = rsqrtf(tot / 128.0f + 1e-6f);
    float xn  = v * rn * w[tid];

    int   i    = tid >> 1;
    float freq = __expf(-(float)i * (logf(10000.0f) / 64.0f));
    float s, c;
    __sincosf((float)t * freq, &s, &c);
    float p = __shfl_xor(xn, 1, 32);
    float r = (tid & 1) ? (p * s + xn * c) : (xn * c - p * s);
    out[((size_t)((size_t)b * HH + h) * TT + t) * HD + tid] = (bf16)r;
}

// ======================================================================
// V prep: [b,t,h,hd] f32 -> [b,h,hd,t] bf16 (transposed for PV B-frags)
// ======================================================================
__global__ __launch_bounds__(128)
void v_prep(const float* __restrict__ in, bf16* __restrict__ out) {
    const int idx = blockIdx.x;
    const int h = idx % HH;
    const int t = (idx / HH) % TT;
    const int b = idx / (HH * TT);
    const int tid = threadIdx.x;
    float v = in[((size_t)((size_t)b * TT + t) * HH + h) * HD + tid];
    out[((size_t)((size_t)b * HH + h) * HD + tid) * TT + t] = (bf16)v;
}

// ======================================================================
// Flash attention (causal). Q,K: [b,h,t,hd] bf16; V: [b,h,hd,t] bf16.
// ctx out: [b,t,h*hd] bf16. Block = (b*h, 128 q-rows), 8 waves x 16 rows.
// K/V tiles double-buffered in LDS via async copies.
// ======================================================================
__global__ __launch_bounds__(256)
void flash_attn(const bf16* __restrict__ Q, const bf16* __restrict__ Kc,
                const bf16* __restrict__ Vt, bf16* __restrict__ ctx) {
    __shared__ bf16 Kt[2][32][128];   // [buf][key][hd]
    __shared__ bf16 Vs[2][128][32];   // [buf][hd][key]
    __shared__ bf16 Pl[8][16][32];    // per-wave P scratch [row][key]

    const int bh   = blockIdx.x;      // b*H + h
    const int qBlk = blockIdx.y;
    const int b    = bh / HH;
    const int h    = bh % HH;
    const int tid  = threadIdx.x;
    const int lane = tid & 31;
    const int wave = tid >> 5;
    const int l15  = lane & 15;
    const int half = lane >> 4;
    const int qr0  = qBlk * 128 + wave * 16;
    const int kbA  = half * 8;
    const int kbB  = half * 8;

    const size_t baseQK = (size_t)bh * TT * HD;

    // loader thread mapping for K/V tiles
    const int krow = tid >> 3, kcol = (tid & 7) * 16;
    const int vhd  = tid >> 1, vcol = (tid & 1) * 16;

    // Q fragments stay resident (16 rows x 128 hd)
    FragU qf[4];
    const bf16* qrow = Q + baseQK + (size_t)(qr0 + l15) * HD;
    for (int c = 0; c < 4; ++c) {
        qf[c].h[0] = *(const v8bf*)(qrow + c * 32 + kbA);
        qf[c].h[1] = *(const v8bf*)(qrow + c * 32 + kbA + 16);
    }

    v8f O[8];
    for (int j = 0; j < 8; ++j) O[j] = {};
    float mrow[8], lrow[8];
    for (int r = 0; r < 8; ++r) { mrow[r] = -1e30f; lrow[r] = 0.0f; }

    const float scale  = 0.08838834764831845f;  // 1/sqrt(128)
    const int   nTiles = qBlk * 4 + 4;          // causal: keys up to qBlk*128+127

    // prologue: stage key-tile 0 into buffer 0
    {
        const bf16* kg = Kc + baseQK + (size_t)krow * HD + kcol;
        async_b128(&Kt[0][krow][kcol],     kg);
        async_b128(&Kt[0][krow][kcol + 8], kg + 8);
        const bf16* vg = Vt + ((size_t)bh * HD + vhd) * TT + vcol;
        async_b128(&Vs[0][vhd][vcol],     vg);
        async_b128(&Vs[0][vhd][vcol + 8], vg + 8);
    }

    for (int kt = 0; kt < nTiles; ++kt) {
        const int s0   = kt * 32;
        const int bufI = kt & 1;
        wait_async0();
        __syncthreads();

        if (kt + 1 < nTiles) {    // stage next tile (overlaps compute)
            const int nb = bufI ^ 1;
            const int s1 = s0 + 32;
            const bf16* kg = Kc + baseQK + (size_t)(s1 + krow) * HD + kcol;
            async_b128(&Kt[nb][krow][kcol],     kg);
            async_b128(&Kt[nb][krow][kcol + 8], kg + 8);
            const bf16* vg = Vt + ((size_t)bh * HD + vhd) * TT + s1 + vcol;
            async_b128(&Vs[nb][vhd][vcol],     vg);
            async_b128(&Vs[nb][vhd][vcol + 8], vg + 8);
        }

        // S = Q * K^T (two 16-key column blocks), mask, row max
        float sval[2][8], mnew[8];
        for (int r = 0; r < 8; ++r) mnew[r] = mrow[r];
        for (int nb = 0; nb < 2; ++nb) {
            v8f S = {};
            const int key = nb * 16 + l15;
            for (int c = 0; c < 4; ++c) {
                FragU bfr;
                bfr.h[0] = *(const v8bf*)&Kt[bufI][key][c * 32 + kbB];
                bfr.h[1] = *(const v8bf*)&Kt[bufI][key][c * 32 + kbB + 16];
                S = wmma_bf16(qf[c], bfr, S);
            }
            const int ts = s0 + nb * 16 + l15;
            for (int r = 0; r < 8; ++r) {
                const int tq = qr0 + r + half * 8;
                float sv = S[r] * scale;
                if (ts > tq) sv = -1e30f;
                sval[nb][r] = sv;
                float rm = sv;
                rm = fmaxf(rm, __shfl_xor(rm, 1, 32));
                rm = fmaxf(rm, __shfl_xor(rm, 2, 32));
                rm = fmaxf(rm, __shfl_xor(rm, 4, 32));
                rm = fmaxf(rm, __shfl_xor(rm, 8, 32));
                mnew[r] = fmaxf(mnew[r], rm);
            }
        }
        float alpha[8];
        for (int r = 0; r < 8; ++r) alpha[r] = __expf(mrow[r] - mnew[r]);

        // P = exp(S - mnew) -> LDS (for A-frag reshape); row sums
        for (int nb = 0; nb < 2; ++nb)
            for (int r = 0; r < 8; ++r) {
                float p = __expf(sval[nb][r] - mnew[r]);
                Pl[wave][r + half * 8][nb * 16 + l15] = (bf16)p;
                float rs = p;
                rs += __shfl_xor(rs, 1, 32);
                rs += __shfl_xor(rs, 2, 32);
                rs += __shfl_xor(rs, 4, 32);
                rs += __shfl_xor(rs, 8, 32);
                if (nb == 0) lrow[r] = lrow[r] * alpha[r] + rs;
                else         lrow[r] += rs;
            }
        for (int r = 0; r < 8; ++r) mrow[r] = mnew[r];

        for (int j = 0; j < 8; ++j)
            for (int r = 0; r < 8; ++r) O[j][r] *= alpha[r];

        __syncthreads();  // make P visible across lanes

        FragU pf;
        pf.h[0] = *(const v8bf*)&Pl[wave][l15][kbA];
        pf.h[1] = *(const v8bf*)&Pl[wave][l15][kbA + 16];
        for (int j = 0; j < 8; ++j) {
            FragU vf;
            const int hd = j * 16 + l15;
            vf.h[0] = *(const v8bf*)&Vs[bufI][hd][kbB];
            vf.h[1] = *(const v8bf*)&Vs[bufI][hd][kbB + 16];
            O[j] = wmma_bf16(pf, vf, O[j]);
        }
    }

    // normalize and write ctx [b,t,h*HD]
    for (int j = 0; j < 8; ++j)
        for (int r = 0; r < 8; ++r) {
            const int tq = qr0 + r + half * 8;
            float o = O[j][r] / lrow[r];
            ctx[((size_t)b * TT + tq) * (HH * HD) + h * HD + j * 16 + l15] = (bf16)o;
        }
}

// ======================================================================
// launcher
// ======================================================================
extern "C" void kernel_launch(void* const* d_in, const int* in_sizes, int n_in,
                              void* d_out, int out_size, void* d_ws, size_t ws_size,
                              hipStream_t stream) {
    (void)in_sizes; (void)n_in; (void)out_size; (void)ws_size;
    const float* x  = (const float*)d_in[0];
    const float* Wq = (const float*)d_in[1];
    const float* Wk = (const float*)d_in[2];
    const float* Wv = (const float*)d_in[3];
    const float* Wo = (const float*)d_in[4];
    const float* qw = (const float*)d_in[5];
    const float* kw = (const float*)d_in[6];
    float* out = (float*)d_out;

    const int M = BB * TT;                 // 4096
    const size_t nX = (size_t)M * DD;      // 8M elems
    const size_t nW = (size_t)DD * DD;     // 4M elems
    const size_t nHead = (size_t)BB * HH * TT * HD;  // 8M elems

    char* ws = (char*)d_ws;
    size_t off = 0;
    bf16* x_bf   = (bf16*)(ws + off); off += nX * sizeof(bf16);
    bf16* Wq_bf  = (bf16*)(ws + off); off += nW * sizeof(bf16);
    bf16* Wk_bf  = (bf16*)(ws + off); off += nW * sizeof(bf16);
    bf16* Wv_bf  = (bf16*)(ws + off); off += nW * sizeof(bf16);
    bf16* Wo_bf  = (bf16*)(ws + off); off += nW * sizeof(bf16);
    float* tmp32 = (float*)(ws + off); off += nX * sizeof(float);   // reused q/k/v
    bf16* q_bf   = (bf16*)(ws + off); off += nHead * sizeof(bf16);
    bf16* k_bf   = (bf16*)(ws + off); off += nHead * sizeof(bf16);
    bf16* v_bf   = (bf16*)(ws + off); off += nHead * sizeof(bf16);
    bf16* ctx_bf = (bf16*)(ws + off); off += nX * sizeof(bf16);

    // 1) casts
    cast_f32_bf16<<<(int)((nX + 255) / 256), 256, 0, stream>>>(x,  x_bf,  (int)nX);
    cast_f32_bf16<<<(int)((nW + 255) / 256), 256, 0, stream>>>(Wq, Wq_bf, (int)nW);
    cast_f32_bf16<<<(int)((nW + 255) / 256), 256, 0, stream>>>(Wk, Wk_bf, (int)nW);
    cast_f32_bf16<<<(int)((nW + 255) / 256), 256, 0, stream>>>(Wv, Wv_bf, (int)nW);
    cast_f32_bf16<<<(int)((nW + 255) / 256), 256, 0, stream>>>(Wo, Wo_bf, (int)nW);

    dim3 gemmGrid(DD / 128, M / 128);      // (16, 32)
    dim3 rowGrid(BB * TT * HH);            // 65536

    // 2) Q/K/V projections + norm/rope/permute
    gemm_bf16_wmma<<<gemmGrid, 256, 0, stream>>>(x_bf, Wq_bf, tmp32, M, DD, DD);
    rmsnorm_rope<<<rowGrid, 128, 0, stream>>>(tmp32, qw, q_bf);
    gemm_bf16_wmma<<<gemmGrid, 256, 0, stream>>>(x_bf, Wk_bf, tmp32, M, DD, DD);
    rmsnorm_rope<<<rowGrid, 128, 0, stream>>>(tmp32, kw, k_bf);
    gemm_bf16_wmma<<<gemmGrid, 256, 0, stream>>>(x_bf, Wv_bf, tmp32, M, DD, DD);
    v_prep<<<rowGrid, 128, 0, stream>>>(tmp32, v_bf);

    // 3) flash attention
    dim3 attnGrid(BB * HH, TT / 128);      // (32, 16)
    flash_attn<<<attnGrid, 256, 0, stream>>>(q_bf, k_bf, v_bf, ctx_bf);

    // 4) output projection -> f32 out
    gemm_bf16_wmma<<<gemmGrid, 256, 0, stream>>>(ctx_bf, Wo_bf, out, M, DD, DD);
}
